// MLPTorsoSetTxAttention_78709570666913
// MI455X (gfx1250) — matrix operand order repaired and checked by reading, not compiled, not run on hardware
//
#include <hip/hip_runtime.h>

typedef __attribute__((ext_vector_type(16))) __bf16 bf16x16;
typedef __attribute__((ext_vector_type(8)))  __bf16 bf16x8;
typedef __attribute__((ext_vector_type(8)))  float  f32x8;

#define TOK   16      // tokens per block (one WMMA M-tile)
#define BLK   256     // 8 wave32 waves
#define XD    200     // input feature dim per token
#define CATD  352     // concat dim (K of f1)
#define HIDD  128     // hidden dim
#define BIGN  (-1.0e9f)

// ---------------- LDS layout (bytes, all 256-aligned) ----------------
#define OFF_W1   0          // bf16 [128][352]  f1 weights transposed (n-major)
#define OFF_W2   90112      // bf16 [128][128]  f2 weights transposed
#define OFF_CAT  122880     // bf16 [16][352]   concat activations (A of f1)
#define OFF_TOK  134144     // f32  [16][5][24] nbr tokens (through stx block)
#define OFF_EGO  141824     // f32  [16][64]    ego embedding (f32 copy for egoq)
#define OFF_SCR  145920     // f32  scratch pool, 9280 floats, phase-aliased
#define OFF_MSK  183040     // f32  mask[80], safe[80], cnt[16], has[16]
#define SMEM_BYTES 183808
// scratch pool offsets (in floats)
#define SC_Y  0      // 1920
#define SC_Q  1920   // 1920
#define SC_K  3840   // 1920
#define SC_V  5760   // 1920
#define SC_S  7680   // 1600

struct Params {
  const float* x;
  const float *ego_w,*ego_b,*pair_w,*pair_b,*ter_w,*ter_b,*tok_w,*tok_b;
  const float *ln1_s,*ln1_b,*wq,*bq,*wk,*bk,*wv,*bv,*wo,*bo,*ln2_s,*ln2_b;
  const float *m1_w,*m1_b,*m2_w,*m2_b;
  const float *egoq_w,*egoq_b,*plnq_s,*plnq_b,*plnkv_s,*plnkv_b;
  const float *pwq,*pbq,*pwk,*pbk,*pwv,*pbv,*pwo,*pbo;
  const float *f1_w,*f1_b,*lf1_s,*lf1_b,*f2_w,*f2_b,*lf2_s,*lf2_b;
  float* out;
};

__device__ __forceinline__ float gelu_t(float v) {
  float c = 0.7978845608028654f * (v + 0.044715f * v * v * v);
  return 0.5f * v * (1.0f + tanhf(c));
}

__device__ __forceinline__ __bf16 tobf(float f) {
  unsigned u = __builtin_bit_cast(unsigned, f);
  u += 0x7fffu + ((u >> 16) & 1u);              // round-to-nearest-even
  unsigned short h = (unsigned short)(u >> 16);
  return __builtin_bit_cast(__bf16, h);
}

// LayerNorm over 24 elements: o = (v-mu)*rsqrt(var+eps)*s + b
__device__ __forceinline__ void ln24(const float* v, const float* s, const float* b, float* o) {
  float mu = 0.f;
#pragma unroll
  for (int j = 0; j < 24; ++j) mu += v[j];
  mu *= (1.0f / 24.0f);
  float var = 0.f;
#pragma unroll
  for (int j = 0; j < 24; ++j) { float d = v[j] - mu; var += d * d; }
  var *= (1.0f / 24.0f);
  float rs = rsqrtf(var + 1e-6f);
#pragma unroll
  for (int j = 0; j < 24; ++j) o[j] = (v[j] - mu) * rs * s[j] + b[j];
}

__launch_bounds__(BLK)
__global__ void torso_fused(Params p) {
  extern __shared__ char smem[];
  __bf16* sW1  = (__bf16*)(smem + OFF_W1);
  __bf16* sW2  = (__bf16*)(smem + OFF_W2);
  __bf16* sCat = (__bf16*)(smem + OFF_CAT);
  float*  sTok = (float*)(smem + OFF_TOK);
  float*  sEgo = (float*)(smem + OFF_EGO);
  float*  SCR  = (float*)(smem + OFF_SCR);
  float*  sMask = (float*)(smem + OFF_MSK);
  float*  sSafe = sMask + 80;
  float*  sCnt  = sSafe + 80;
  float*  sHas  = sCnt + 16;

  const int tid   = threadIdx.x;
  const int gbase = blockIdx.x * TOK;

  // prefetch this block's input rows (global_prefetch_b8)
  for (int t = tid; t < TOK; t += BLK)
    __builtin_prefetch(p.x + (size_t)(gbase + t) * XD, 0, 0);

  // ---- Phase 0: stage f1/f2 weights into LDS transposed [n][k], bf16 ----
  for (int i = tid; i < CATD * HIDD; i += BLK) {          // f1_w is (352,128) row-major
    int k = i >> 7, n = i & 127;
    sW1[n * CATD + k] = tobf(p.f1_w[i]);
  }
  for (int i = tid; i < HIDD * HIDD; i += BLK) {
    int k = i >> 7, n = i & 127;
    sW2[n * HIDD + k] = tobf(p.f2_w[i]);
  }

  // ---- Phase 1: masks ----
  for (int i = tid; i < TOK * 5; i += BLK) {
    int t = i / 5, s = i % 5;
    const float* px = p.x + (size_t)(gbase + t) * XD + 34 + s * 15;
    float m = 0.f;
#pragma unroll
    for (int q = 0; q < 15; ++q) m = fmaxf(m, fabsf(px[q]));
    sMask[i] = (m > 1e-6f) ? 1.f : 0.f;
  }
  __syncthreads();
  for (int t = tid; t < TOK; t += BLK) {
    float c = 0.f;
    for (int s = 0; s < 5; ++s) c += sMask[t * 5 + s];
    sCnt[t] = fmaxf(c, 1.f);
    sHas[t] = (c > 0.f) ? 1.f : 0.f;
  }
  __syncthreads();
  for (int i = tid; i < TOK * 5; i += BLK) {
    int t = i / 5, s = i % 5;
    sSafe[i] = (sHas[t] > 0.f) ? sMask[i] : (s == 0 ? 1.f : 0.f);
  }

  // ---- Phase 1b: embeddings -> sCat (bf16) + sEgo/sTok (f32) ----
  for (int i = tid; i < TOK * 64; i += BLK) {             // ego 34->64
    int t = i >> 6, o = i & 63;
    const float* px = p.x + (size_t)(gbase + t) * XD;
    float a = p.ego_b[o];
#pragma unroll
    for (int q = 0; q < 34; ++q) a += px[q] * p.ego_w[q * 64 + o];
    a = gelu_t(a);
    sEgo[i] = a;
    sCat[t * CATD + o] = tobf(a);
  }
  for (int i = tid; i < TOK * 32; i += BLK) {             // pair 10->32
    int t = i >> 5, o = i & 31;
    const float* px = p.x + (size_t)(gbase + t) * XD + 109;
    float a = p.pair_b[o];
#pragma unroll
    for (int q = 0; q < 10; ++q) a += px[q] * p.pair_w[q * 32 + o];
    sCat[t * CATD + 64 + o] = tobf(gelu_t(a));
  }
  for (int i = tid; i < TOK * 64; i += BLK) {             // ter 81->64
    int t = i >> 6, o = i & 63;
    const float* px = p.x + (size_t)(gbase + t) * XD + 119;
    float a = p.ter_b[o];
    for (int q = 0; q < 81; ++q) a += px[q] * p.ter_w[q * 64 + o];
    sCat[t * CATD + 96 + o] = tobf(gelu_t(a));
  }
  for (int i = tid; i < TOK * 120; i += BLK) {            // nbr tokens 15->24 (masked)
    int t = i / 120, r = i % 120, s = r / 24, j = r % 24;
    const float* px = p.x + (size_t)(gbase + t) * XD + 34 + s * 15;
    float a = p.tok_b[j];
#pragma unroll
    for (int q = 0; q < 15; ++q) a += px[q] * p.tok_w[q * 24 + j];
    sTok[i] = gelu_t(a) * sMask[t * 5 + s];
  }
  __syncthreads();

  float* sY = SCR + SC_Y;
  float* sQ = SCR + SC_Q;
  float* sK = SCR + SC_K;
  float* sV = SCR + SC_V;
  float* sS = SCR + SC_S;

  // ---- Phase 2: stx LN1 + q/k/v ----
  for (int i = tid; i < TOK * 5; i += BLK)
    ln24(&sTok[i * 24], p.ln1_s, p.ln1_b, &sY[i * 24]);
  __syncthreads();
  for (int i = tid; i < TOK * 120; i += BLK) {
    int ts = i / 24, hf = i % 24;
    const float* y = &sY[ts * 24];
    float aq = p.bq[hf], ak = p.bk[hf], av = p.bv[hf];
#pragma unroll
    for (int d = 0; d < 24; ++d) {
      float yv = y[d];
      aq += yv * p.wq[d * 24 + hf];
      ak += yv * p.wk[d * 24 + hf];
      av += yv * p.wv[d * 24 + hf];
    }
    sQ[i] = aq; sK[i] = ak; sV[i] = av;
  }
  __syncthreads();

  // ---- Phase 3: stx attention (4 heads, 5x5, head dim 6) ----
  const float scl = 0.4082482904638631f;   // 1/sqrt(6)
  for (int i = tid; i < TOK * 100; i += BLK) {
    int t = i / 100, r = i % 100, h = r / 25, r2 = r % 25, qi = r2 / 5, kj = r2 % 5;
    float a = 0.f;
#pragma unroll
    for (int f = 0; f < 6; ++f)
      a += sQ[(t * 5 + qi) * 24 + h * 6 + f] * sK[(t * 5 + kj) * 24 + h * 6 + f];
    a *= scl;
    if (!(sSafe[t * 5 + qi] > 0.f && sSafe[t * 5 + kj] > 0.f)) a = BIGN;
    sS[i] = a;
  }
  __syncthreads();
  for (int i = tid; i < TOK * 20; i += BLK) {             // softmax over 5 keys
    float* q5 = &sS[i * 5];
    float m = q5[0];
#pragma unroll
    for (int j = 1; j < 5; ++j) m = fmaxf(m, q5[j]);
    float sum = 0.f;
#pragma unroll
    for (int j = 0; j < 5; ++j) { float e = __expf(q5[j] - m); q5[j] = e; sum += e; }
    float inv = 1.f / sum;
#pragma unroll
    for (int j = 0; j < 5; ++j) q5[j] *= inv;
  }
  __syncthreads();
  for (int i = tid; i < TOK * 120; i += BLK) {            // attn out -> sQ (dead)
    int t = i / 120, r = i % 120, qi = r / 24, hf = r % 24, h = hf / 6;
    float a = 0.f;
#pragma unroll
    for (int j = 0; j < 5; ++j)
      a += sS[((t * 4 + h) * 5 + qi) * 5 + j] * sV[(t * 5 + j) * 24 + hf];
    sQ[(t * 5 + qi) * 24 + hf] = a;
  }
  __syncthreads();
  for (int i = tid; i < TOK * 120; i += BLK) {            // wo + masked residual
    int ts = i / 24, dd = i % 24;
    float a = p.bo[dd];
#pragma unroll
    for (int hf = 0; hf < 24; ++hf) a += sQ[ts * 24 + hf] * p.wo[hf * 24 + dd];
    if (sMask[ts] > 0.f) sTok[i] += a;
  }
  __syncthreads();

  // ---- Phase 4: stx MLP 24->48->24 ----
  for (int i = tid; i < TOK * 5; i += BLK)
    ln24(&sTok[i * 24], p.ln2_s, p.ln2_b, &sY[i * 24]);
  __syncthreads();
  float* sH48 = SCR + SC_Q;                               // 3840 floats (Q+K dead)
  for (int i = tid; i < TOK * 5 * 48; i += BLK) {
    int ts = i / 48, o = i % 48;
    float a = p.m1_b[o];
#pragma unroll
    for (int d = 0; d < 24; ++d) a += sY[ts * 24 + d] * p.m1_w[d * 48 + o];
    sH48[i] = gelu_t(a);
  }
  __syncthreads();
  for (int i = tid; i < TOK * 120; i += BLK) {
    int ts = i / 24, o = i % 24;
    float a = p.m2_b[o];
#pragma unroll
    for (int d = 0; d < 48; ++d) a += sH48[ts * 48 + d] * p.m2_w[d * 24 + o];
    if (sMask[ts] > 0.f) sTok[i] += a;
  }
  __syncthreads();

  // ---- Phase 5: pooling (flat / mean / max / ego-query attention) ----
  for (int i = tid; i < TOK * 120; i += BLK) {            // nbr_flat
    int t = i / 120, r = i % 120;
    sCat[t * CATD + 160 + r] = tobf(sTok[i]);
  }
  for (int i = tid; i < TOK * 24; i += BLK) {             // mean + max
    int t = i / 24, j = i % 24;
    float sm = 0.f, mx = BIGN;
#pragma unroll
    for (int s = 0; s < 5; ++s) {
      float v = sTok[(t * 5 + s) * 24 + j];
      sm += v;
      if (sMask[t * 5 + s] > 0.f) mx = fmaxf(mx, v);
    }
    sCat[t * CATD + 280 + j] = tobf(sm / sCnt[t]);
    sCat[t * CATD + 304 + j] = tobf(sHas[t] > 0.f ? mx : 0.f);
  }
  float* sQp = SCR + SC_Q;       // 16*24  LN'd ego query
  float* sKv = SCR + SC_Y;       // 1920   LN'd kv
  float* sKp = SCR + SC_K;
  float* sVp = SCR + SC_V;
  float* sSp = SCR + SC_S;       // 16*4*5 scores
  float* sOp = SCR + SC_S + 640; // 16*24  q-projection / reuse
  for (int t = tid; t < TOK; t += BLK) {                  // ego_q + LN(q)
    float qv[24], qo[24];
#pragma unroll
    for (int j = 0; j < 24; ++j) {
      float a = p.egoq_b[j];
      for (int d = 0; d < 64; ++d) a += sEgo[t * 64 + d] * p.egoq_w[d * 24 + j];
      qv[j] = a;
    }
    ln24(qv, p.plnq_s, p.plnq_b, qo);
#pragma unroll
    for (int j = 0; j < 24; ++j) sQp[t * 24 + j] = qo[j];
  }
  for (int i = tid; i < TOK * 5; i += BLK)                // LN(kv)
    ln24(&sTok[i * 24], p.plnkv_s, p.plnkv_b, &sKv[i * 24]);
  __syncthreads();
  for (int i = tid; i < TOK * 24; i += BLK) {             // q projection -> sOp
    int t = i / 24, hf = i % 24;
    float a = p.pbq[hf];
#pragma unroll
    for (int d = 0; d < 24; ++d) a += sQp[t * 24 + d] * p.pwq[d * 24 + hf];
    sOp[i] = a;
  }
  for (int i = tid; i < TOK * 120; i += BLK) {            // k/v projections
    int ts = i / 24, hf = i % 24;
    float ak = p.pbk[hf], av = p.pbv[hf];
#pragma unroll
    for (int d = 0; d < 24; ++d) {
      float y = sKv[ts * 24 + d];
      ak += y * p.pwk[d * 24 + hf];
      av += y * p.pwv[d * 24 + hf];
    }
    sKp[i] = ak; sVp[i] = av;
  }
  __syncthreads();
  for (int i = tid; i < TOK * 20; i += BLK) {             // scores (t,h,j)
    int t = i / 20, r = i % 20, h = r / 5, j = r % 5;
    float a = 0.f;
#pragma unroll
    for (int f = 0; f < 6; ++f)
      a += sOp[t * 24 + h * 6 + f] * sKp[(t * 5 + j) * 24 + h * 6 + f];
    a *= scl;
    if (!(sSafe[t * 5 + j] > 0.f)) a = BIGN;
    sSp[i] = a;
  }
  __syncthreads();
  for (int i = tid; i < TOK * 4; i += BLK) {              // softmax
    float* q5 = &sSp[i * 5];
    float m = q5[0];
#pragma unroll
    for (int j = 1; j < 5; ++j) m = fmaxf(m, q5[j]);
    float sum = 0.f;
#pragma unroll
    for (int j = 0; j < 5; ++j) { float e = __expf(q5[j] - m); q5[j] = e; sum += e; }
    float inv = 1.f / sum;
#pragma unroll
    for (int j = 0; j < 5; ++j) q5[j] *= inv;
  }
  __syncthreads();
  for (int i = tid; i < TOK * 24; i += BLK) {             // weighted v -> sQp (dead)
    int t = i / 24, hf = i % 24, h = hf / 6;
    float a = 0.f;
#pragma unroll
    for (int j = 0; j < 5; ++j)
      a += sSp[(t * 4 + h) * 5 + j] * sVp[(t * 5 + j) * 24 + hf];
    sQp[i] = a;
  }
  __syncthreads();
  for (int i = tid; i < TOK * 24; i += BLK) {             // pool wo + gate -> sCat
    int t = i / 24, dd = i % 24;
    float a = p.pbo[dd];
#pragma unroll
    for (int hf = 0; hf < 24; ++hf) a += sQp[t * 24 + hf] * p.pwo[hf * 24 + dd];
    sCat[t * CATD + 328 + dd] = tobf(sHas[t] > 0.f ? a : 0.f);
  }
  __syncthreads();

  // ---- Phase 6: f1 via WMMA bf16 (M=16, K=352, N=128; one N-tile per wave) ----
  float*  sH1  = SCR + SC_Y;                 // 2048 f32 (Y+Q regions, dead)
  __bf16* sH1b = (__bf16*)(SCR + SC_V);      // 2048 bf16 (V region, dead)
  float*  sMu  = SCR + SC_S;
  float*  sRs  = SCR + SC_S + 16;
  const int wv_  = tid >> 5;                 // wave id = N-tile
  const int ln_  = tid & 31;
  const int mrow = ln_ & 15;                 // A row / B column / D column-in-tile
  const int hi   = ln_ >> 4;                 // lane half (K phase)
  {
    f32x8 acc = {0.f,0.f,0.f,0.f,0.f,0.f,0.f,0.f};
#pragma unroll
    for (int kt = 0; kt < 11; ++kt) {
      const __bf16* pa = sCat + mrow * CATD + kt * 32 + hi * 16;
      bf16x16 A = *(const bf16x16*)pa;
      const __bf16* pb = sW1 + (wv_ * 16 + mrow) * CATD + kt * 32 + hi * 16;
      bf16x16 Bf = *(const bf16x16*)pb;
      acc = __builtin_amdgcn_wmma_f32_16x16x32_bf16(false, A, false, Bf, (short)0, acc,
                                                    false, false);
    }
    const int n = wv_ * 16 + mrow;
#pragma unroll
    for (int r = 0; r < 8; ++r) {            // D layout: VGPR r, lane-half -> M
      int mm = r + hi * 8;
      sH1[mm * HIDD + n] = gelu_t(acc[r] + p.f1_b[n]);
    }
  }
  __syncthreads();
  for (int t = tid; t < TOK; t += BLK) {     // LN stats over 128
    float mu = 0.f;
    for (int n = 0; n < HIDD; ++n) mu += sH1[t * HIDD + n];
    mu *= (1.0f / HIDD);
    float var = 0.f;
    for (int n = 0; n < HIDD; ++n) { float d = sH1[t * HIDD + n] - mu; var += d * d; }
    sMu[t] = mu;
    sRs[t] = rsqrtf(var * (1.0f / HIDD) + 1e-6f);
  }
  __syncthreads();
  for (int i = tid; i < TOK * HIDD; i += BLK) {
    int t = i >> 7, n = i & 127;
    sH1b[i] = tobf((sH1[i] - sMu[t]) * sRs[t] * p.lf1_s[n] + p.lf1_b[n]);
  }
  __syncthreads();

  // ---- Phase 7: f2 via WMMA bf16 (M=16, K=128, N=128) ----
  {
    f32x8 acc = {0.f,0.f,0.f,0.f,0.f,0.f,0.f,0.f};
#pragma unroll
    for (int kt = 0; kt < 4; ++kt) {
      const __bf16* pa = sH1b + mrow * HIDD + kt * 32 + hi * 16;
      bf16x16 A = *(const bf16x16*)pa;
      const __bf16* pb = sW2 + (wv_ * 16 + mrow) * HIDD + kt * 32 + hi * 16;
      bf16x16 Bf = *(const bf16x16*)pb;
      acc = __builtin_amdgcn_wmma_f32_16x16x32_bf16(false, A, false, Bf, (short)0, acc,
                                                    false, false);
    }
    const int n = wv_ * 16 + mrow;
#pragma unroll
    for (int r = 0; r < 8; ++r) {
      int mm = r + hi * 8;
      sH1[mm * HIDD + n] = gelu_t(acc[r] + p.f2_b[n]);
    }
  }
  __syncthreads();
  for (int t = tid; t < TOK; t += BLK) {
    float mu = 0.f;
    for (int n = 0; n < HIDD; ++n) mu += sH1[t * HIDD + n];
    mu *= (1.0f / HIDD);
    float var = 0.f;
    for (int n = 0; n < HIDD; ++n) { float d = sH1[t * HIDD + n] - mu; var += d * d; }
    sMu[t] = mu;
    sRs[t] = rsqrtf(var * (1.0f / HIDD) + 1e-6f);
  }
  __syncthreads();
  for (int i = tid; i < TOK * HIDD; i += BLK) {           // final LN -> global (coalesced)
    int t = i >> 7, n = i & 127;
    float v = (sH1[i] - sMu[t]) * sRs[t] * p.lf2_s[n] + p.lf2_b[n];
    p.out[(size_t)(gbase + t) * HIDD + n] = v;
  }
}

extern "C" void kernel_launch(void* const* d_in, const int* in_sizes, int n_in,
                              void* d_out, int out_size, void* d_ws, size_t ws_size,
                              hipStream_t stream) {
  (void)n_in; (void)d_ws; (void)ws_size; (void)out_size;
  Params p;
  int i = 0;
  p.x      = (const float*)d_in[i++];
  p.ego_w  = (const float*)d_in[i++];  p.ego_b  = (const float*)d_in[i++];
  p.pair_w = (const float*)d_in[i++];  p.pair_b = (const float*)d_in[i++];
  p.ter_w  = (const float*)d_in[i++];  p.ter_b  = (const float*)d_in[i++];
  p.tok_w  = (const float*)d_in[i++];  p.tok_b  = (const float*)d_in[i++];
  p.ln1_s  = (const float*)d_in[i++];  p.ln1_b  = (const float*)d_in[i++];
  p.wq     = (const float*)d_in[i++];  p.bq     = (const float*)d_in[i++];
  p.wk     = (const float*)d_in[i++];  p.bk     = (const float*)d_in[i++];
  p.wv     = (const float*)d_in[i++];  p.bv     = (const float*)d_in[i++];
  p.wo     = (const float*)d_in[i++];  p.bo     = (const float*)d_in[i++];
  p.ln2_s  = (const float*)d_in[i++];  p.ln2_b  = (const float*)d_in[i++];
  p.m1_w   = (const float*)d_in[i++];  p.m1_b   = (const float*)d_in[i++];
  p.m2_w   = (const float*)d_in[i++];  p.m2_b   = (const float*)d_in[i++];
  p.egoq_w = (const float*)d_in[i++];  p.egoq_b = (const float*)d_in[i++];
  p.plnq_s = (const float*)d_in[i++];  p.plnq_b = (const float*)d_in[i++];
  p.plnkv_s= (const float*)d_in[i++];  p.plnkv_b= (const float*)d_in[i++];
  p.pwq    = (const float*)d_in[i++];  p.pbq    = (const float*)d_in[i++];
  p.pwk    = (const float*)d_in[i++];  p.pbk    = (const float*)d_in[i++];
  p.pwv    = (const float*)d_in[i++];  p.pbv    = (const float*)d_in[i++];
  p.pwo    = (const float*)d_in[i++];  p.pbo    = (const float*)d_in[i++];
  p.f1_w   = (const float*)d_in[i++];  p.f1_b   = (const float*)d_in[i++];
  p.lf1_s  = (const float*)d_in[i++];  p.lf1_b  = (const float*)d_in[i++];
  p.f2_w   = (const float*)d_in[i++];  p.f2_b   = (const float*)d_in[i++];
  p.lf2_s  = (const float*)d_in[i++];  p.lf2_b  = (const float*)d_in[i++];
  p.out    = (float*)d_out;

  long tokens = (long)in_sizes[0] / XD;     // 131072 for the reference shapes
  int grid = (int)(tokens / TOK);

  hipFuncSetAttribute(reinterpret_cast<const void*>(torso_fused),
                      hipFuncAttributeMaxDynamicSharedMemorySize, SMEM_BYTES);
  torso_fused<<<grid, BLK, SMEM_BYTES, stream>>>(p);
}
